// Block_86655260164584
// MI455X (gfx1250) — compile-verified, matrix operand order checked by reading, not run on hardware
//
#include <hip/hip_runtime.h>
#include <hip/hip_bf16.h>

typedef __attribute__((ext_vector_type(16))) _Float16 v16h;
typedef __attribute__((ext_vector_type(8)))  _Float16 v8h;
typedef __attribute__((ext_vector_type(4)))  _Float16 v4h;
typedef __attribute__((ext_vector_type(8)))  float    v8f;

static __device__ __forceinline__ v8f wmma16(v16h a, v16h b, v8f c) {
  return __builtin_amdgcn_wmma_f32_16x16x32_f16(false, a, false, b, (short)0, c, false, false);
}
static __device__ __forceinline__ v16h pack16(v8h lo, v8h hi) {
  return __builtin_shufflevector(lo, hi, 0,1,2,3,4,5,6,7,8,9,10,11,12,13,14,15);
}

// ---------------------------------------------------------------------------
// fp32 -> f16 conversion (vectorized x4); totals are multiples of 4
// ---------------------------------------------------------------------------
__global__ __launch_bounds__(256)
void cvt16_kernel(const float* __restrict__ in, _Float16* __restrict__ out, int total) {
  const int i = (blockIdx.x * 256 + threadIdx.x) * 4;
  if (i >= total) return;
  const float4 v = *(const float4*)(in + i);
  v4h o; o[0] = (_Float16)v.x; o[1] = (_Float16)v.y; o[2] = (_Float16)v.z; o[3] = (_Float16)v.w;
  *(v4h*)(out + i) = o;
}

// ---------------------------------------------------------------------------
// Epilogues: 0=identity, 1=relu^2, 2=sigmoid, 3=gelu(exact)
// ---------------------------------------------------------------------------
template<int EPI>
static __device__ __forceinline__ float apply_epi(float x) {
  if constexpr (EPI == 1) { float t = x > 0.f ? x : 0.f; return t * t; }
  else if constexpr (EPI == 2) { return 1.f / (1.f + __expf(-x)); }
  else if constexpr (EPI == 3) { return 0.5f * x * (1.f + erff(x * 0.70710678118654752f)); }
  else return x;
}

// ---------------------------------------------------------------------------
// WMMA GEMM, all-f16 operands: out[M,N] = epi( A[M,K] @ W[N,K]^T )
// W tile (64 x 32) staged in LDS, double-buffered, shared by all 8 waves.
// Requirements: M % 256 == 0, K % 32 == 0.  N guarded in 16-wide steps.
// ---------------------------------------------------------------------------
template<int EPI, int OH>
__global__ __launch_bounds__(256)
void gemm_kernel(const _Float16* __restrict__ A, const _Float16* __restrict__ W,
                 void* __restrict__ Op, int M, int N, int K) {
  const int tid  = threadIdx.x;
  const int wave = tid >> 5;
  const int lane = tid & 31;
  const int hlf  = lane >> 4;
  const int r    = lane & 15;
  const int m0 = blockIdx.x * 256 + wave * 32;
  const int n0 = blockIdx.y * 64;

  __shared__ __attribute__((aligned(16))) _Float16 wtile[2][64 * 32];
  const int srow = tid >> 2;            // 0..63
  const int scol = (tid & 3) * 8;       // 0,8,16,24

  auto stage = [&](int buf, int k0) {
    if (n0 + srow < N) {
      const v8h w = *(const v8h*)(W + (size_t)(n0 + srow) * K + k0 + scol);
      *(v8h*)(&wtile[buf][srow * 32 + scol]) = w;
    }
  };

  v8f acc[2][4];
  #pragma unroll
  for (int mi = 0; mi < 2; ++mi)
    #pragma unroll
    for (int nj = 0; nj < 4; ++nj)
      #pragma unroll
      for (int e = 0; e < 8; ++e) acc[mi][nj][e] = 0.f;

  stage(0, 0);
  __syncthreads();

  int it = 0;
  for (int k0 = 0; k0 < K; k0 += 32, ++it) {
    const int cur = it & 1;
    if (k0 + 32 < K) stage(cur ^ 1, k0 + 32);

    const _Float16* p0 = A + (size_t)(m0 + r) * K + k0 + hlf * 8;
    const _Float16* p1 = A + (size_t)(m0 + 16 + r) * K + k0 + hlf * 8;
    const v16h a0 = pack16(*(const v8h*)p0, *(const v8h*)(p0 + 16));
    const v16h a1 = pack16(*(const v8h*)p1, *(const v8h*)(p1 + 16));
    if (k0 + 32 < K) __builtin_prefetch(p0 + 32, 0, 0);

    #pragma unroll
    for (int nj = 0; nj < 4; ++nj) {
      const int nb = n0 + nj * 16;
      if (nb >= N) break;
      const _Float16* lp = &wtile[cur][(nj * 16 + r) * 32 + hlf * 8];
      const v16h bf = pack16(*(const v8h*)lp, *(const v8h*)(lp + 16));
      acc[0][nj] = wmma16(a0, bf, acc[0][nj]);
      acc[1][nj] = wmma16(a1, bf, acc[1][nj]);
    }
    __syncthreads();
  }

  #pragma unroll
  for (int nj = 0; nj < 4; ++nj) {
    const int nb = n0 + nj * 16;
    if (nb >= N) break;
    const int n = nb + r;
    #pragma unroll
    for (int mi = 0; mi < 2; ++mi)
      #pragma unroll
      for (int v = 0; v < 8; ++v) {
        const int m = m0 + mi * 16 + v + 8 * hlf;
        const float val = apply_epi<EPI>(acc[mi][nj][v]);
        if constexpr (OH) ((_Float16*)Op)[(size_t)m * N + n] = (_Float16)val;
        else              ((float*)Op)[(size_t)m * N + n] = val;
      }
  }
}

// ---------------------------------------------------------------------------
// RMSNorm over last dim C (one block per row), f32 in -> f16 out
// ---------------------------------------------------------------------------
__global__ __launch_bounds__(256)
void rmsnorm16_kernel(const float* __restrict__ x, _Float16* __restrict__ o, int C) {
  const size_t row = blockIdx.x;
  const float* xr = x + row * (size_t)C;
  float ss = 0.f;
  for (int i = threadIdx.x; i < C; i += 256) { float v = xr[i]; ss += v * v; }
  #pragma unroll
  for (int off = 16; off; off >>= 1) ss += __shfl_xor(ss, off, 32);
  __shared__ float red[8];
  if ((threadIdx.x & 31) == 0) red[threadIdx.x >> 5] = ss;
  __syncthreads();
  float tot = 0.f;
  #pragma unroll
  for (int i = 0; i < 8; ++i) tot += red[i];
  const float rs = rsqrtf(tot / (float)C + 1e-6f);
  _Float16* orow = o + row * (size_t)C;
  for (int i = threadIdx.x; i < C; i += 256) orow[i] = (_Float16)(xr[i] * rs);
}

// ---------------------------------------------------------------------------
// QKV post: RoPE + head RMSNorm*1.2 for q,k; v + 3*sigmoid(xn[:12]@gate)*ve.
// grid = (B*T, 24) blocks of 64 threads.  Outputs f16 heads.
// qh: [B,H,T,64]   kh: [B,KV,T,64]   vt: [B,KV,64,T]  (V TRANSPOSED)
// ---------------------------------------------------------------------------
__global__ __launch_bounds__(64)
void qkvpost_kernel(const float* __restrict__ q_raw, const float* __restrict__ k_raw,
                    const float* __restrict__ v_raw, const _Float16* __restrict__ xn,
                    const float* __restrict__ ve, const float* __restrict__ cosb,
                    const float* __restrict__ sinb, const float* __restrict__ wgate,
                    _Float16* __restrict__ qh, _Float16* __restrict__ kh,
                    _Float16* __restrict__ vt, int T) {
  const int bt = blockIdx.x;
  const int b = bt / T, t = bt % T;
  const int u = blockIdx.y;
  const int lane = threadIdx.x;
  __shared__ float red[2];

  if (u < 20) {
    const float* src;
    _Float16* dst;
    if (u < 16) { src = q_raw + (size_t)bt * 1024 + u * 64;
                  dst = qh + (((size_t)(b * 16 + u)) * T + t) * 64; }
    else        { const int kv = u - 16;
                  src = k_raw + (size_t)bt * 256 + kv * 64;
                  dst = kh + (((size_t)(b * 4 + kv)) * T + t) * 64; }
    const int j = lane & 31;
    const float c = cosb[(size_t)t * 32 + j], s = sinb[(size_t)t * 32 + j];
    const float x1 = src[j], x2 = src[j + 32];
    const float val = (lane < 32) ? (x1 * c + x2 * s) : (-x1 * s + x2 * c);
    float ss = val * val;
    #pragma unroll
    for (int off = 16; off; off >>= 1) ss += __shfl_xor(ss, off, 32);
    if ((lane & 31) == 0) red[lane >> 5] = ss;
    __syncthreads();
    const float rs = rsqrtf((red[0] + red[1]) * (1.f / 64.f) + 1e-6f) * 1.2f;
    dst[lane] = (_Float16)(val * rs);
  } else {
    const int kv = u - 20;
    float g = 0.f;
    #pragma unroll
    for (int i = 0; i < 12; ++i) g += (float)xn[(size_t)bt * 1024 + i] * wgate[kv * 12 + i];
    g = 3.f / (1.f + __expf(-g));
    const size_t idx = (size_t)bt * 256 + kv * 64 + lane;
    // transposed store: [b][kv][hd=lane][t]
    vt[(((size_t)(b * 4 + kv)) * 64 + lane) * T + t] = (_Float16)(v_raw[idx] + g * ve[idx]);
  }
}

// ---------------------------------------------------------------------------
// Flash attention with refine-conv term.  grid = (B*H, T/128), block = 256.
// Wave w handles 16 query rows; keys tiled 32 wide; causal.  f16 in, f16 out.
// V consumed transposed [B,KV,64,T] -> B-fragments are two b128 loads.
// Refine conv is branchless: 10 rows x 3 clamped+masked columns per lane,
// reused across the 8 accumulator rows (9 FMAs each).
// ---------------------------------------------------------------------------
__global__ __launch_bounds__(256)
void attn_kernel(const _Float16* __restrict__ qh, const _Float16* __restrict__ kh,
                 const _Float16* __restrict__ vt, const float* __restrict__ prev,
                 const float* __restrict__ refine_w, const float* __restrict__ alpha_p,
                 _Float16* __restrict__ y, int T) {
  constexpr int H = 16, KV = 4, HD = 64;
  const int bh = blockIdx.x;
  const int b = bh / H, h = bh % H;
  const int wave = threadIdx.x >> 5;
  const int lane = threadIdx.x & 31;
  const int hlf = lane >> 4, r = lane & 15;
  const int qb = (blockIdx.y * 8 + wave) * 16;
  const float alpha = alpha_p[0];

  float rw[9];
  #pragma unroll
  for (int i = 0; i < 9; ++i) rw[i] = refine_w[h * 9 + i];

  const _Float16* qbase = qh + ((size_t)(b * H + h)) * T * HD;
  const _Float16* kbase = kh + ((size_t)(b * KV + (h >> 2))) * T * HD;
  const _Float16* vbase = vt + ((size_t)(b * KV + (h >> 2))) * HD * T;
  const float*    pbase = prev + ((size_t)(b * H + h)) * T * T;

  v16h aQ[2];
  {
    const _Float16* qp = qbase + (size_t)(qb + r) * HD + hlf * 8;
    aQ[0] = pack16(*(const v8h*)qp,        *(const v8h*)(qp + 16));
    aQ[1] = pack16(*(const v8h*)(qp + 32), *(const v8h*)(qp + 48));
  }

  float mstat[8], lstat[8];
  v8f O[4];
  #pragma unroll
  for (int v = 0; v < 8; ++v) { mstat[v] = -1e30f; lstat[v] = 0.f; }
  #pragma unroll
  for (int j = 0; j < 4; ++j)
    #pragma unroll
    for (int e = 0; e < 8; ++e) O[j][e] = 0.f;

  __shared__ __attribute__((aligned(16))) _Float16 lds_p[8 * 16 * 32];
  const int wbase = wave * 512;
  const int qrow0 = qb + 8 * hlf;   // first accumulator row for this lane-half

  for (int kb = 0; kb <= qb + 15; kb += 32) {
    // S = Q @ K^T  (16 x 32 tile, two 16x16 fragments)
    v8f S[2];
    #pragma unroll
    for (int sb = 0; sb < 2; ++sb) {
      #pragma unroll
      for (int e = 0; e < 8; ++e) S[sb][e] = 0.f;
      const _Float16* kp = kbase + (size_t)(kb + sb * 16 + r) * HD + hlf * 8;
      S[sb] = wmma16(aQ[0], pack16(*(const v8h*)kp,        *(const v8h*)(kp + 16)), S[sb]);
      S[sb] = wmma16(aQ[1], pack16(*(const v8h*)(kp + 32), *(const v8h*)(kp + 48)), S[sb]);
    }

    // scale + refine conv (branchless, row-reused) + causal mask
    #pragma unroll
    for (int sb = 0; sb < 2; ++sb) {
      const int kj = kb + sb * 16 + r;
      const float cm0 = (kj > 0) ? 1.f : 0.f;
      const float cm2 = (kj < T - 1) ? 1.f : 0.f;
      const int c0 = (kj > 0) ? kj - 1 : 0;
      const int c2 = (kj < T - 1) ? kj + 1 : kj;
      float colL[10], colC[10], colR[10];
      #pragma unroll
      for (int i = 0; i < 10; ++i) {
        const int rr = qrow0 - 1 + i;
        const float rm = ((unsigned)rr < (unsigned)T) ? 1.f : 0.f;
        const int rrc = (rr < 0) ? 0 : ((rr >= T) ? T - 1 : rr);
        const float* pr = pbase + (size_t)rrc * T;
        colL[i] = rm * cm0 * pr[c0];
        colC[i] = rm * pr[kj];
        colR[i] = rm * cm2 * pr[c2];
      }
      #pragma unroll
      for (int v = 0; v < 8; ++v) {
        float acc = 0.f;
        #pragma unroll
        for (int di = 0; di < 3; ++di)
          acc += colL[v + di] * rw[di * 3] + colC[v + di] * rw[di * 3 + 1]
               + colR[v + di] * rw[di * 3 + 2];
        const int qi = qrow0 + v;
        float s = S[sb][v] * 0.125f + alpha * acc;
        if (kj > qi) s = -1e30f;
        S[sb][v] = s;
      }
    }

    // online softmax (row stats via 16-lane shuffles), stage P to LDS
    #pragma unroll
    for (int v = 0; v < 8; ++v) {
      float rmax = fmaxf(S[0][v], S[1][v]);
      #pragma unroll
      for (int off = 1; off < 16; off <<= 1) rmax = fmaxf(rmax, __shfl_xor(rmax, off, 16));
      const float mnew = fmaxf(mstat[v], rmax);
      const float scl = __expf(mstat[v] - mnew);
      mstat[v] = mnew;
      lstat[v] *= scl;
      #pragma unroll
      for (int j = 0; j < 4; ++j) O[j][v] *= scl;
      const float p0 = __expf(S[0][v] - mnew);
      const float p1 = __expf(S[1][v] - mnew);
      float rsum = p0 + p1;
      #pragma unroll
      for (int off = 1; off < 16; off <<= 1) rsum += __shfl_xor(rsum, off, 16);
      lstat[v] += rsum;
      const int m = v + 8 * hlf;
      lds_p[wbase + m * 32 + r]      = (_Float16)p0;
      lds_p[wbase + m * 32 + 16 + r] = (_Float16)p1;
    }

    // O += P @ V   (V transposed: contiguous b128 B-fragment loads)
    v16h aP;
    {
      const _Float16* pp = lds_p + wbase + r * 32 + hlf * 8;
      aP = pack16(*(const v8h*)pp, *(const v8h*)(pp + 16));
    }
    #pragma unroll
    for (int j = 0; j < 4; ++j) {
      const _Float16* vp = vbase + (size_t)(j * 16 + r) * T + kb + hlf * 8;
      const v16h bv = pack16(*(const v8h*)vp, *(const v8h*)(vp + 16));
      O[j] = wmma16(aP, bv, O[j]);
    }
  }

  #pragma unroll
  for (int v = 0; v < 8; ++v) {
    const float inv = 1.f / lstat[v];
    #pragma unroll
    for (int j = 0; j < 4; ++j) O[j][v] *= inv;
  }
  #pragma unroll
  for (int j = 0; j < 4; ++j)
    #pragma unroll
    for (int v = 0; v < 8; ++v) {
      const int qi = qb + v + 8 * hlf;
      const int hd = j * 16 + r;
      y[((size_t)(b * T + qi)) * 1024 + h * 64 + hd] = (_Float16)O[j][v];
    }
}

// ---------------------------------------------------------------------------
// Channel-attention depthwise conv along T + exact GELU -> f16 (GEMM input)
// ---------------------------------------------------------------------------
__global__ __launch_bounds__(256)
void caconv_kernel(const float* __restrict__ h, const float* __restrict__ cw,
                   _Float16* __restrict__ o, int T, int CH, int total) {
  const int idx = blockIdx.x * 256 + threadIdx.x;
  if (idx >= total) return;
  const int c = idx % CH;
  const int t = (idx / CH) % T;
  const float ctr = h[idx];
  const float lft = (t > 0)     ? h[idx - CH] : 0.f;
  const float rgt = (t < T - 1) ? h[idx + CH] : 0.f;
  const float v = ctr + 0.1f * (lft * cw[c * 3] + ctr * cw[c * 3 + 1] + rgt * cw[c * 3 + 2]);
  o[idx] = (_Float16)(0.5f * v * (1.f + erff(v * 0.70710678118654752f)));
}

// FFN depthwise conv step (f16 ping-pong): o = in + 0.1*conv_T(in)
__global__ __launch_bounds__(256)
void ffnconv_kernel(const _Float16* __restrict__ in, const float* __restrict__ cw,
                    _Float16* __restrict__ o, int T, int F, int total) {
  const int idx = blockIdx.x * 256 + threadIdx.x;
  if (idx >= total) return;
  const int f = idx % F;
  const int t = (idx / F) % T;
  const float ctr = (float)in[idx];
  const float lft = (t > 0)     ? (float)in[idx - F] : 0.f;
  const float rgt = (t < T - 1) ? (float)in[idx + F] : 0.f;
  o[idx] = (_Float16)(ctr + 0.1f * (lft * cw[f * 3] + ctr * cw[f * 3 + 1] + rgt * cw[f * 3 + 2]));
}

// x1 = x + attn_proj*(1+0.1*tanh(ca1));  writes f32 and f16 copies
__global__ __launch_bounds__(256)
void x1_kernel(const float* __restrict__ x, const float* __restrict__ ap,
               const float* __restrict__ ca1, float* __restrict__ o,
               _Float16* __restrict__ o16, int total) {
  const int idx = blockIdx.x * 256 + threadIdx.x;
  if (idx >= total) return;
  const float v = x[idx] + ap[idx] * (1.f + 0.1f * tanhf(ca1[idx]));
  o[idx] = v;
  o16[idx] = (_Float16)v;
}

__global__ __launch_bounds__(256)
void vitdot_kernel(const float* __restrict__ g32, const float* __restrict__ w2,
                   float* __restrict__ vit0, int ntok) {
  const int i = blockIdx.x * 256 + threadIdx.x;
  if (i >= ntok) return;
  float s = 0.f;
  #pragma unroll
  for (int j = 0; j < 32; ++j) s += g32[(size_t)i * 32 + j] * w2[j];
  vit0[i] = 1.f / (1.f + __expf(-s));
}

__global__ __launch_bounds__(256)
void vitsmooth_kernel(const float* __restrict__ vit0, float* __restrict__ vit,
                      int T, int ntok) {
  const int i = blockIdx.x * 256 + threadIdx.x;
  if (i >= ntok) return;
  const int t = i % T;
  const int base = i - t;
  float s = 0.f;
  #pragma unroll
  for (int d = -2; d <= 2; ++d) {
    const int tt = t + d;
    if ((unsigned)tt < (unsigned)T) s += vit0[base + tt];
  }
  const float v = 0.7f * vit0[i] + 0.3f * (s * 0.2f);
  vit[i] = (v > 0.3f) ? v : 0.1f * v;
}

__global__ __launch_bounds__(256)
void final_kernel(const float* __restrict__ x1, const float* __restrict__ mlp,
                  const float* __restrict__ g, const float* __restrict__ ca2,
                  const float* __restrict__ vit, float* __restrict__ out,
                  int C, int total) {
  const int idx = blockIdx.x * 256 + threadIdx.x;
  if (idx >= total) return;
  const int tok = idx / C;
  out[idx] = x1[idx] + mlp[idx] * g[idx] * (1.f + 0.1f * tanhf(ca2[idx])) * vit[tok];
}

// ---------------------------------------------------------------------------
extern "C" void kernel_launch(void* const* d_in, const int* in_sizes, int n_in,
                              void* d_out, int out_size, void* d_ws, size_t ws_size,
                              hipStream_t stream) {
  (void)in_sizes; (void)n_in; (void)out_size; (void)ws_size;
  constexpr int B = 4, T = 1024, C = 1024, H = 16, KV = 4, HD = 64, FFN = 4096, CA = 32;
  constexpr int BT = B * T;

  const float* x       = (const float*)d_in[0];
  const float* ve      = (const float*)d_in[1];
  const float* cosb    = (const float*)d_in[2];
  const float* sinb    = (const float*)d_in[3];
  const float* prev    = (const float*)d_in[4];
  const float* w_q     = (const float*)d_in[5];
  const float* w_k     = (const float*)d_in[6];
  const float* w_v     = (const float*)d_in[7];
  const float* w_o     = (const float*)d_in[8];
  const float* w_gate  = (const float*)d_in[9];
  const float* rf_w    = (const float*)d_in[10];
  const float* rf_a    = (const float*)d_in[11];
  const float* ca_pi   = (const float*)d_in[12];
  const float* ca_cw   = (const float*)d_in[13];
  const float* ca_po   = (const float*)d_in[14];
  const float* ffn_in  = (const float*)d_in[15];
  const float* ffn_cw  = (const float*)d_in[16];
  const float* ffn_out = (const float*)d_in[17];
  const float* ffn_g   = (const float*)d_in[18];
  const float* vit_w1  = (const float*)d_in[19];
  const float* vit_w2  = (const float*)d_in[20];
  float* out = (float*)d_out;

  float* wsf = (float*)d_ws;
  size_t off = 0;
  auto allocf = [&](size_t n) -> float* {
    float* p = wsf + off;
    off += (n + 63) & ~(size_t)63;
    return p;
  };
  auto alloch = [&](size_t n) -> _Float16* {   // n halfs
    return (_Float16*)allocf((n + 1) / 2);
  };

  // fp32 working buffers
  float* q_raw     = allocf((size_t)BT * C);
  float* k_raw     = allocf((size_t)BT * KV * HD);
  float* v_raw     = allocf((size_t)BT * KV * HD);
  float* attn_proj = allocf((size_t)BT * C);
  float* ca_h      = allocf((size_t)BT * CA);
  float* ca1       = allocf((size_t)BT * C);
  float* x1        = allocf((size_t)BT * C);
  float* vit0      = allocf(BT);
  float* vitf      = allocf(BT);
  // f16 working buffers
  _Float16* x16    = alloch((size_t)BT * C);
  _Float16* xn16   = alloch((size_t)BT * C);
  _Float16* qh     = alloch((size_t)BT * C);
  _Float16* kh     = alloch((size_t)BT * KV * HD);
  _Float16* vt     = alloch((size_t)BT * KV * HD);   // transposed [B,KV,64,T]
  _Float16* y16    = alloch((size_t)BT * C);
  _Float16* cag16  = alloch((size_t)BT * CA);
  _Float16* x1_16  = alloch((size_t)BT * C);
  _Float16* hA     = alloch((size_t)BT * FFN);
  _Float16* hB     = alloch((size_t)BT * FFN);
  // f16 weights (converted once)
  _Float16* w_q16   = alloch((size_t)C * C);
  _Float16* w_k16   = alloch((size_t)KV * HD * C);
  _Float16* w_v16   = alloch((size_t)KV * HD * C);
  _Float16* w_o16   = alloch((size_t)C * C);
  _Float16* ca_pi16 = alloch((size_t)CA * C);
  _Float16* ca_po16 = alloch((size_t)C * CA);
  _Float16* fin16   = alloch((size_t)FFN * C);
  _Float16* fout16  = alloch((size_t)C * FFN);
  _Float16* fg16    = alloch((size_t)C * C);
  _Float16* vw1_16  = alloch((size_t)CA * C);
  // reuse
  float* gbuf  = q_raw;      // q_raw dead after qkvpost
  float* mlp   = attn_proj;  // dead after x1
  float* ca2   = ca1;        // dead after x1
  float* g32   = ca_h;       // dead after ca2 conv
  _Float16* xm16 = xn16;     // xn16 dead after QKV + gate

  const dim3 blk(256);
  const int MB = BT / 256;  // 16
  auto CVT = [&](const float* s, _Float16* d, int n) {
    cvt16_kernel<<<dim3((n / 4 + 255) / 256), blk, 0, stream>>>(s, d, n);
  };

  // --- one-time f16 conversions (weights + x) ---
  CVT(x, x16, BT * C);
  CVT(w_q, w_q16, C * C);
  CVT(w_k, w_k16, KV * HD * C);
  CVT(w_v, w_v16, KV * HD * C);
  CVT(w_o, w_o16, C * C);
  CVT(ca_pi, ca_pi16, CA * C);
  CVT(ca_po, ca_po16, C * CA);
  CVT(ffn_in, fin16, FFN * C);
  CVT(ffn_out, fout16, C * FFN);
  CVT(ffn_g, fg16, C * C);
  CVT(vit_w1, vw1_16, CA * C);

  // --- attention path ---
  rmsnorm16_kernel<<<BT, blk, 0, stream>>>(x, xn16, C);
  gemm_kernel<0,0><<<dim3(MB, C / 64), blk, 0, stream>>>(xn16, w_q16, q_raw, BT, C, C);
  gemm_kernel<0,0><<<dim3(MB, (KV * HD) / 64), blk, 0, stream>>>(xn16, w_k16, k_raw, BT, KV * HD, C);
  gemm_kernel<0,0><<<dim3(MB, (KV * HD) / 64), blk, 0, stream>>>(xn16, w_v16, v_raw, BT, KV * HD, C);
  qkvpost_kernel<<<dim3(BT, H + 2 * KV), dim3(64), 0, stream>>>(
      q_raw, k_raw, v_raw, xn16, ve, cosb, sinb, w_gate, qh, kh, vt, T);
  attn_kernel<<<dim3(B * H, T / 128), blk, 0, stream>>>(qh, kh, vt, prev, rf_w, rf_a, y16, T);
  gemm_kernel<0,0><<<dim3(MB, C / 64), blk, 0, stream>>>(y16, w_o16, attn_proj, BT, C, C);

  // --- ca1 + residual into x1 ---
  gemm_kernel<0,0><<<dim3(MB, 1), blk, 0, stream>>>(x16, ca_pi16, ca_h, BT, CA, C);
  caconv_kernel<<<dim3((BT * CA) / 256), blk, 0, stream>>>(ca_h, ca_cw, cag16, T, CA, BT * CA);
  gemm_kernel<0,0><<<dim3(MB, C / 64), blk, 0, stream>>>(cag16, ca_po16, ca1, BT, C, CA);
  x1_kernel<<<dim3((BT * C) / 256), blk, 0, stream>>>(x, attn_proj, ca1, x1, x1_16, BT * C);

  // --- FFN ---
  rmsnorm16_kernel<<<BT, blk, 0, stream>>>(x1, xm16, C);
  gemm_kernel<1,1><<<dim3(MB, FFN / 64), blk, 0, stream>>>(xm16, fin16, hA, BT, FFN, C);
  const int ftot = BT * FFN;
  ffnconv_kernel<<<dim3(ftot / 256), blk, 0, stream>>>(hA, ffn_cw, hB, T, FFN, ftot);
  ffnconv_kernel<<<dim3(ftot / 256), blk, 0, stream>>>(hB, ffn_cw, hA, T, FFN, ftot);
  ffnconv_kernel<<<dim3(ftot / 256), blk, 0, stream>>>(hA, ffn_cw, hB, T, FFN, ftot);
  ffnconv_kernel<<<dim3(ftot / 256), blk, 0, stream>>>(hB, ffn_cw, hA, T, FFN, ftot);
  gemm_kernel<0,0><<<dim3(MB, C / 64), blk, 0, stream>>>(hA, fout16, mlp, BT, C, FFN);
  gemm_kernel<2,0><<<dim3(MB, C / 64), blk, 0, stream>>>(xm16, fg16, gbuf, BT, C, C);

  // --- ca2 ---
  gemm_kernel<0,0><<<dim3(MB, 1), blk, 0, stream>>>(x1_16, ca_pi16, ca_h, BT, CA, C);
  caconv_kernel<<<dim3((BT * CA) / 256), blk, 0, stream>>>(ca_h, ca_cw, cag16, T, CA, BT * CA);
  gemm_kernel<0,0><<<dim3(MB, C / 64), blk, 0, stream>>>(cag16, ca_po16, ca2, BT, C, CA);

  // --- vit gate ---
  gemm_kernel<3,0><<<dim3(MB, 1), blk, 0, stream>>>(x1_16, vw1_16, g32, BT, CA, C);
  vitdot_kernel<<<dim3(BT / 256), blk, 0, stream>>>(g32, vit_w2, vit0, BT);
  vitsmooth_kernel<<<dim3(BT / 256), blk, 0, stream>>>(vit0, vitf, T, BT);

  // --- final combine ---
  final_kernel<<<dim3((BT * C) / 256), blk, 0, stream>>>(x1, mlp, gbuf, ca2, vitf, out, C, BT * C);
}
